// RelationalLayer_6966436954182
// MI455X (gfx1250) — matrix operand order repaired and checked by reading, not compiled, not run on hardware
//
#include <hip/hip_runtime.h>
#include <hip/hip_bf16.h>
#include <math.h>

// ---------------------------------------------------------------------------
// Relation Network forward, fused for MI455X (gfx1250, wave32, WMMA bf16).
//   B=32, D=64, K=26, QST=256, G=[512,512,512,512,512,512,28], AGG=4
// Heavy part: 131072 x (52->512->512->512->512) + sum over 4096 pairs/batch.
// Design: activations LDS-resident (zero HBM traffic for intermediates),
// weights bf16 streamed from L2, v_wmma_f32_16x16x32_bf16 inner loop with
// 2-way B reuse per A fragment (1 DS load per WMMA).
// ---------------------------------------------------------------------------

typedef __attribute__((ext_vector_type(16))) __bf16 v16bf;
typedef __attribute__((ext_vector_type(8)))  float  v8f;

union Frag32 { uint4 u[2]; v16bf v; };

// Workspace layout (bytes)
#define OFF_S   0                      // float S[32][512]        (65536 B)
#define OFF_W0  65536                  // bf16  W0b[512][64]      (65536 B)
#define OFF_W1  131072                 // bf16  W1b[512][512]     (524288 B)
#define OFF_W2  655360                 // bf16  W2b[512][512]
#define OFF_W3  1179648                // bf16  W3b[512][512]     end = 1703936

__device__ __forceinline__ unsigned short f2bf(float f) {
  union { float f; unsigned int u; } a; a.f = f;
  unsigned int r = a.u + 0x7FFFu + ((a.u >> 16) & 1u);   // round-to-nearest-even
  return (unsigned short)(r >> 16);
}

__device__ __forceinline__ v8f wmma_bf16(v16bf a, v16bf b, v8f c) {
  // D = A(16x32 bf16) x B(32x16 bf16) + C(16x16 f32)
  return __builtin_amdgcn_wmma_f32_16x16x32_bf16(
      /*neg_a=*/false, a, /*neg_b=*/false, b,
      /*c_mod=*/(short)0, c, /*reuse_a=*/false, /*reuse_b=*/false);
}

// ---------------------------------------------------------------------------
// Prep: fp32 weights -> bf16 (W0 padded K 52->64), zero aggregation buffer S.
// ---------------------------------------------------------------------------
__global__ void rn_prep(const float* __restrict__ W0, const float* __restrict__ W1,
                        const float* __restrict__ W2, const float* __restrict__ W3,
                        char* __restrict__ ws)
{
  float*          S   = (float*)(ws + OFF_S);
  unsigned short* W0b = (unsigned short*)(ws + OFF_W0);
  unsigned short* W1b = (unsigned short*)(ws + OFF_W1);
  unsigned short* W2b = (unsigned short*)(ws + OFF_W2);
  unsigned short* W3b = (unsigned short*)(ws + OFF_W3);

  const int i0 = blockIdx.x * blockDim.x + threadIdx.x;
  const int stride = gridDim.x * blockDim.x;

  for (int i = i0; i < 32 * 512; i += stride) S[i] = 0.0f;
  for (int i = i0; i < 512 * 64; i += stride) {
    int n = i >> 6, k = i & 63;
    W0b[i] = f2bf(k < 52 ? W0[n * 52 + k] : 0.0f);
  }
  for (int i = i0; i < 512 * 512; i += stride) {
    W1b[i] = f2bf(W1[i]);
    W2b[i] = f2bf(W2[i]);
    W3b[i] = f2bf(W3[i]);
  }
}

// ---------------------------------------------------------------------------
// Tile writeback: bias + ReLU, then either bf16 store to LDS Hout or
// column-sum reduction into LDS f32 Spart (last layer).
// ---------------------------------------------------------------------------
__device__ __forceinline__ void tile_out(int nt, int hi, int ln,
                                         v8f acc0, v8f acc1, v8f acc2, v8f acc3,
                                         const float* __restrict__ bias,
                                         unsigned short* __restrict__ Hout,
                                         float* __restrict__ Spart)
{
  const float bn  = bias[nt * 16 + ln];
  const int   col = nt * 16 + ln;
  if (Hout) {
    #pragma unroll
    for (int v = 0; v < 8; ++v) {
      const int m = hi * 8 + v;                 // D layout: VGPR v, lane-half hi
      Hout[(m +  0) * 512 + col] = f2bf(fmaxf(acc0[v] + bn, 0.0f));
      Hout[(m + 16) * 512 + col] = f2bf(fmaxf(acc1[v] + bn, 0.0f));
      Hout[(m + 32) * 512 + col] = f2bf(fmaxf(acc2[v] + bn, 0.0f));
      Hout[(m + 48) * 512 + col] = f2bf(fmaxf(acc3[v] + bn, 0.0f));
    }
  } else {
    float cs = 0.0f;
    #pragma unroll
    for (int v = 0; v < 8; ++v) {
      cs += fmaxf(acc0[v] + bn, 0.0f) + fmaxf(acc1[v] + bn, 0.0f)
          + fmaxf(acc2[v] + bn, 0.0f) + fmaxf(acc3[v] + bn, 0.0f);
    }
    atomicAdd(&Spart[col], cs);                 // ds_add_f32
  }
}

// ---------------------------------------------------------------------------
// One g-layer for a 64-row tile, activations LDS-resident.
//   Hin : LDS bf16 [64][kin]    W : global bf16 [512][kin]
//   Hout: LDS bf16 [64][512] (or null for the last layer)
//   Spart: LDS f32 [512] column partial sums (last layer only)
// 8 waves; wave w owns N-tiles {w, w+8, w+16, w+24} processed as 2 groups of
// 2 so each A fragment (DS load) feeds 2 WMMAs -> 1 ds_load_b128 per wmma.
// ---------------------------------------------------------------------------
__device__ __forceinline__ void g_layer(const unsigned short* __restrict__ Hin, int kin,
                                        const unsigned short* __restrict__ W,
                                        const float* __restrict__ bias,
                                        unsigned short* __restrict__ Hout,
                                        float* __restrict__ Spart)
{
  const int tid  = threadIdx.x;
  const int lane = tid & 31;
  const int wave = tid >> 5;
  const int hi   = lane >> 4;     // lane half (K-interleave select)
  const int ln   = lane & 15;     // A: row M ; B: col N ; D: col N
  const int ksteps = kin >> 5;

  const unsigned short* arow = Hin + ln * kin + hi * 8;

  #pragma unroll
  for (int g = 0; g < 2; ++g) {
    const int ntA = wave + g * 16;   // group handles n-tiles ntA and ntA+8
    const int ntB = ntA + 8;

    v8f accA0 = {0,0,0,0,0,0,0,0};
    v8f accA1 = accA0, accA2 = accA0, accA3 = accA0;
    v8f accB0 = accA0, accB1 = accA0, accB2 = accA0, accB3 = accA0;

    // B fragments: lane ln holds weight row (B column) nt*16+ln,
    // elems 0..7 at K = ks*32 + hi*8, elems 8..15 at +16 (matches A).
    const unsigned short* wrowA = W + (size_t)(ntA * 16 + ln) * kin + hi * 8;
    const unsigned short* wrowB = W + (size_t)(ntB * 16 + ln) * kin + hi * 8;

    for (int ks = 0; ks < ksteps; ++ks) {
      const int k0 = ks * 32;
      Frag32 bA, bB;
      bA.u[0] = *(const uint4*)(wrowA + k0);
      bA.u[1] = *(const uint4*)(wrowA + k0 + 16);
      bB.u[0] = *(const uint4*)(wrowB + k0);
      bB.u[1] = *(const uint4*)(wrowB + k0 + 16);

      // A fragments: elems 0..7 at K = k0 + hi*8, elems 8..15 at +16.
      Frag32 a0, a1, a2, a3;
      a0.u[0] = *(const uint4*)(arow + k0);
      a0.u[1] = *(const uint4*)(arow + k0 + 16);
      a1.u[0] = *(const uint4*)(arow + 16 * kin + k0);
      a1.u[1] = *(const uint4*)(arow + 16 * kin + k0 + 16);
      a2.u[0] = *(const uint4*)(arow + 32 * kin + k0);
      a2.u[1] = *(const uint4*)(arow + 32 * kin + k0 + 16);
      a3.u[0] = *(const uint4*)(arow + 48 * kin + k0);
      a3.u[1] = *(const uint4*)(arow + 48 * kin + k0 + 16);

      accA0 = wmma_bf16(a0.v, bA.v, accA0);
      accB0 = wmma_bf16(a0.v, bB.v, accB0);
      accA1 = wmma_bf16(a1.v, bA.v, accA1);
      accB1 = wmma_bf16(a1.v, bB.v, accB1);
      accA2 = wmma_bf16(a2.v, bA.v, accA2);
      accB2 = wmma_bf16(a2.v, bB.v, accB2);
      accA3 = wmma_bf16(a3.v, bA.v, accA3);
      accB3 = wmma_bf16(a3.v, bB.v, accB3);
    }

    tile_out(ntA, hi, ln, accA0, accA1, accA2, accA3, bias, Hout, Spart);
    tile_out(ntB, hi, ln, accB0, accB1, accB2, accB3, bias, Hout, Spart);
  }
}

// ---------------------------------------------------------------------------
// Fused g-MLP over 64 pair rows (fixed p = blockIdx.x, q = 0..63).
// LDS: pairbuf 8KB | HA 64KB | HB 64KB | Spart 2KB  = 141312 B dynamic.
// ---------------------------------------------------------------------------
#define SMEM_BYTES (8192 + 65536 + 65536 + 2048)

__global__ void __launch_bounds__(256)
rn_g(const float* __restrict__ x,                         // [32][64][26]
     const float* __restrict__ b0, const float* __restrict__ b1,
     const float* __restrict__ b2, const float* __restrict__ b3,
     const char* __restrict__ ws_ro, float* __restrict__ S)
{
  extern __shared__ char smem[];
  unsigned short* pairbuf = (unsigned short*)smem;                 // [64][64]
  unsigned short* HA      = (unsigned short*)(smem + 8192);        // [64][512]
  unsigned short* HB      = (unsigned short*)(smem + 8192 + 65536);
  float*          Spart   = (float*)(smem + 8192 + 2 * 65536);     // [512]

  const unsigned short* W0b = (const unsigned short*)(ws_ro + OFF_W0);
  const unsigned short* W1b = (const unsigned short*)(ws_ro + OFF_W1);
  const unsigned short* W2b = (const unsigned short*)(ws_ro + OFF_W2);
  const unsigned short* W3b = (const unsigned short*)(ws_ro + OFF_W3);

  const int tid = threadIdx.x;
  const int b   = blockIdx.y;
  const int p   = blockIdx.x;

  for (int i = tid; i < 512; i += 256) Spart[i] = 0.0f;

  // Layer-0 input tile: row q -> [x[b][q][0..25], x[b][p][0..25], pad]
  const float* xb = x + (size_t)b * 64 * 26;
  const float* xp = xb + p * 26;
  for (int i = tid; i < 64 * 64; i += 256) {
    const int q = i >> 6, c = i & 63;
    float v = 0.0f;
    if (c < 26)      v = xb[q * 26 + c];
    else if (c < 52) v = xp[c - 26];
    pairbuf[i] = f2bf(v);
  }
  __syncthreads();

  g_layer(pairbuf,  64, W0b, b0, HA, nullptr);
  __syncthreads();
  g_layer(HA,      512, W1b, b1, HB, nullptr);
  __syncthreads();
  g_layer(HB,      512, W2b, b2, HA, nullptr);
  __syncthreads();
  g_layer(HA,      512, W3b, b3, (unsigned short*)nullptr, Spart);
  __syncthreads();

  for (int i = tid; i < 512; i += 256)
    atomicAdd(&S[b * 512 + i], Spart[i]);       // global_atomic_add_f32
}

// ---------------------------------------------------------------------------
// Head: h=[S|qst] -> L4(relu) -> L5(relu) -> L6 -> log_softmax.  32 rows only.
// ---------------------------------------------------------------------------
__global__ void __launch_bounds__(256)
rn_head(const float* __restrict__ S, const float* __restrict__ qst,
        const float* __restrict__ W4, const float* __restrict__ b4,
        const float* __restrict__ W5, const float* __restrict__ b5,
        const float* __restrict__ W6, const float* __restrict__ b6,
        float* __restrict__ out)
{
  __shared__ float in0[768];
  __shared__ float h[512];
  __shared__ float logits[28];
  const int b = blockIdx.x, tid = threadIdx.x;

  for (int i = tid; i < 512; i += 256) in0[i] = S[b * 512 + i];
  for (int i = tid; i < 256; i += 256) in0[512 + i] = qst[b * 256 + i];
  __syncthreads();

  for (int o = tid; o < 512; o += 256) {        // L4: 768 -> 512
    float a = b4[o];
    const float* w = W4 + (size_t)o * 768;
    for (int k = 0; k < 768; ++k) a += in0[k] * w[k];
    h[o] = fmaxf(a, 0.0f);
  }
  __syncthreads();

  for (int o = tid; o < 512; o += 256) {        // L5: 512 -> 512 (into in0)
    float a = b5[o];
    const float* w = W5 + (size_t)o * 512;
    for (int k = 0; k < 512; ++k) a += h[k] * w[k];
    in0[o] = fmaxf(a, 0.0f);
  }
  __syncthreads();

  for (int o = tid; o < 28; o += 256) {         // L6: 512 -> 28
    float a = b6[o];
    const float* w = W6 + (size_t)o * 512;
    for (int k = 0; k < 512; ++k) a += in0[k] * w[k];
    logits[o] = a;
  }
  __syncthreads();

  if (tid == 0) {                               // log_softmax over 28
    float mx = logits[0];
    for (int o = 1; o < 28; ++o) mx = fmaxf(mx, logits[o]);
    float s = 0.0f;
    for (int o = 0; o < 28; ++o) s += expf(logits[o] - mx);
    const float lse = logf(s) + mx;
    for (int o = 0; o < 28; ++o) out[b * 28 + o] = logits[o] - lse;
  }
}

// ---------------------------------------------------------------------------
extern "C" void kernel_launch(void* const* d_in, const int* in_sizes, int n_in,
                              void* d_out, int out_size, void* d_ws, size_t ws_size,
                              hipStream_t stream) {
  const float* x   = (const float*)d_in[0];
  const float* qst = (const float*)d_in[1];
  const float* W0  = (const float*)d_in[2];  const float* b0 = (const float*)d_in[3];
  const float* W1  = (const float*)d_in[4];  const float* b1 = (const float*)d_in[5];
  const float* W2  = (const float*)d_in[6];  const float* b2 = (const float*)d_in[7];
  const float* W3  = (const float*)d_in[8];  const float* b3 = (const float*)d_in[9];
  const float* W4  = (const float*)d_in[10]; const float* b4 = (const float*)d_in[11];
  const float* W5  = (const float*)d_in[12]; const float* b5 = (const float*)d_in[13];
  const float* W6  = (const float*)d_in[14]; const float* b6 = (const float*)d_in[15];

  char*  ws = (char*)d_ws;
  float* S  = (float*)(ws + OFF_S);

  rn_prep<<<1024, 256, 0, stream>>>(W0, W1, W2, W3, ws);
  rn_g<<<dim3(64, 32), 256, SMEM_BYTES, stream>>>(x, b0, b1, b2, b3, ws, S);
  rn_head<<<32, 256, 0, stream>>>(S, qst, W4, b4, W5, b5, W6, b6, (float*)d_out);
}
